// SelfAttention_24369644437965
// MI455X (gfx1250) — matrix operand order repaired and checked by reading, not compiled, not run on hardware
//
#include <hip/hip_runtime.h>

typedef __attribute__((ext_vector_type(16))) _Float16 v16h;
typedef __attribute__((ext_vector_type(8)))  _Float16 v8h;
typedef __attribute__((ext_vector_type(8)))  float    v8f;

#define NN 8192
#define DD 256

#define KROW 264  // padded halfs per staged K/W row (528 B: conflict-free b128 reads)
#define VROW 40   // padded halfs per staged V row   (80 B:  conflict-free b128 reads)
#define PROW 40   // padded halfs per P-bounce row

static __device__ __forceinline__ v8f wmma_f16(v16h a, v16h b, v8f c) {
  // (neg_a, A, neg_b, B, c_mod, C, reuse_a, reuse_b)
  return __builtin_amdgcn_wmma_f32_16x16x32_f16(false, a, false, b, (short)0, c,
                                                false, false);
}

// Async DMA: 16 bytes per lane, global -> LDS, tracked by ASYNCcnt.
static __device__ __forceinline__ void async_b128(unsigned lds_byte_addr,
                                                  const void* gaddr) {
  asm volatile("global_load_async_to_lds_b128 %0, %1, off"
               :: "v"(lds_byte_addr), "v"(gaddr)
               : "memory");
}

// LDS aperture: low 32 bits of a flat pointer to __shared__ are the LDS offset.
static __device__ __forceinline__ unsigned lds_off(const void* p) {
  return (unsigned)(unsigned long long)p;
}

// ---------------------------------------------------------------------------
// QKV projection, one launch for all three (blockIdx.y selects q/k/v).
// 8 waves/block; the whole 256x256 W is staged once in LDS as f16.
// q/k: compute W @ X^T so each lane's 8 accumulator values are 8 consecutive
//      output columns of one output row -> one packed b128 store per tile.
// v:   compute X @ W^T and store transposed (vT[d][key]); each lane's 8
//      values are 8 consecutive keys of one vT row -> one packed b128 store per tile.
// Dual accumulator chains hide ds_load->wmma latency.
// ---------------------------------------------------------------------------
__global__ __launch_bounds__(256) void qkv_gemm(
    const float* __restrict__ x1, const float* __restrict__ x2,
    const float* __restrict__ x3, const float* __restrict__ Wq,
    const float* __restrict__ Wk, const float* __restrict__ Wv,
    const float* __restrict__ bq, const float* __restrict__ bk,
    const float* __restrict__ bv, _Float16* __restrict__ q,
    _Float16* __restrict__ k, _Float16* __restrict__ vt) {
  __shared__ _Float16 wsm[256][KROW];  // 135168 B (f16 copy of W)

  const float* X; const float* W; const float* bias; _Float16* Y; int tr;
  if (blockIdx.y == 0)      { X = x1; W = Wq; bias = bq; Y = q;  tr = 0; }
  else if (blockIdx.y == 1) { X = x2; W = Wk; bias = bk; Y = k;  tr = 0; }
  else                      { X = x3; W = Wv; bias = bv; Y = vt; tr = 1; }

  const int tid   = threadIdx.x;
  const int lane  = tid & 31;
  const int wave  = tid >> 5;
  const int n     = lane & 15;
  const int hs    = lane >> 4;
  const int khalf = hs * 8;

  // ---- stage W into LDS (one 256-float row per thread, f32 -> f16) ----
  {
    const float* wr = W + (size_t)tid * DD;
    _Float16* ds = &wsm[tid][0];
#pragma unroll
    for (int j = 0; j < DD; j += 8) {
      v8h h;
#pragma unroll
      for (int e = 0; e < 8; ++e) { h[e] = (_Float16)wr[j + e]; }
      *(v8h*)(ds + j) = h;
    }
  }
  __syncthreads();

  const int row0 = (blockIdx.x * 8 + wave) * 16;

  if (!tr) {
    // ================= q/k path: D' = W @ X^T =================
    // B-tiles = X rows (lane n -> x row row0+n), interleaved (table) layout.
    v16h xb[8];
    {
      const float* xrow = X + (size_t)(row0 + n) * DD + khalf;
#pragma unroll
      for (int s = 0; s < 8; ++s) {
#pragma unroll
        for (int j = 0; j < 16; ++j) { xb[s][j] = (_Float16)xrow[s * 32 + (j & 7) + ((j >> 3) << 4)]; }
      }
    }
    for (int ct = 0; ct < 16; ++ct) {
      const float* bp = bias + ct * 16 + hs * 8;  // bias along VGPR axis
      v8f c0, c1;
#pragma unroll
      for (int r = 0; r < 8; ++r) { c0[r] = bp[r]; c1[r] = 0.0f; }

      const _Float16* wr = &wsm[ct * 16 + n][0];  // A row m = lane&15
#pragma unroll
      for (int s = 0; s < 8; s += 2) {
        v16h a0, a1;
        const v8h l0 = *(const v8h*)(wr + s * 32 + khalf);
        const v8h h0 = *(const v8h*)(wr + s * 32 + 16 + khalf);
        const v8h l1 = *(const v8h*)(wr + (s + 1) * 32 + khalf);
        const v8h h1 = *(const v8h*)(wr + (s + 1) * 32 + 16 + khalf);
#pragma unroll
        for (int j = 0; j < 8; ++j) {
          a0[j] = l0[j]; a0[j + 8] = h0[j];
          a1[j] = l1[j]; a1[j + 8] = h1[j];
        }
        c0 = wmma_f16(a0, xb[s], c0);
        c1 = wmma_f16(a1, xb[s + 1], c1);
      }
      // lane n: output row row0+n, cols ct*16 + hs*8 + (0..7)  -> one b128
      v8h hv;
#pragma unroll
      for (int r = 0; r < 8; ++r) { hv[r] = (_Float16)(c0[r] + c1[r]); }
      *(v8h*)(Y + (size_t)(row0 + n) * DD + ct * 16 + hs * 8) = hv;
    }
  } else {
    // ================= v path: D = X @ W^T, stored transposed =================
    v16h xa[8];  // A-tiles (contiguous layout)
    {
      const float* xr = X + (size_t)(row0 + n) * DD;
#pragma unroll
      for (int s = 0; s < 8; ++s) {
        const float* ax = xr + s * 32 + hs * 16;
#pragma unroll
        for (int j = 0; j < 8; ++j) {
          xa[s][j]     = (_Float16)ax[j];
          xa[s][j + 8] = (_Float16)ax[j + 8];
        }
      }
    }
    for (int ct = 0; ct < 16; ++ct) {
      const float bv2 = bias[ct * 16 + n];  // C/D layout: lane holds one column
      v8f c0, c1;
#pragma unroll
      for (int r = 0; r < 8; ++r) { c0[r] = bv2; c1[r] = 0.0f; }

      const _Float16* wr = &wsm[ct * 16 + n][hs * 16];
#pragma unroll
      for (int s = 0; s < 8; s += 2) {
        v16h b0, b1;
        const v8h l0 = *(const v8h*)(wr + s * 32);
        const v8h h0 = *(const v8h*)(wr + s * 32 + 8);
        const v8h l1 = *(const v8h*)(wr + (s + 1) * 32);
        const v8h h1 = *(const v8h*)(wr + (s + 1) * 32 + 8);
#pragma unroll
        for (int j = 0; j < 8; ++j) {
          b0[j] = l0[j]; b0[j + 8] = h0[j];
          b1[j] = l1[j]; b1[j + 8] = h1[j];
        }
        c0 = wmma_f16(xa[s], b0, c0);
        c1 = wmma_f16(xa[s + 1], b1, c1);
      }
      // lane n: vT row ct*16+n, cols row0 + hs*8 + (0..7) contiguous -> b128
      v8h hv;
#pragma unroll
      for (int r = 0; r < 8; ++r) { hv[r] = (_Float16)(c0[r] + c1[r]); }
      *(v8h*)(Y + (size_t)(ct * 16 + n) * NN + row0 + hs * 8) = hv;
    }
  }
}

// ---------------------------------------------------------------------------
// Flash attention. 4 waves/block, each wave owns 16 query rows (64 rows/block).
// Per 32-key step the block async-DMAs the K tile (32x256 f16) and V^T tile
// (256x32 f16) into double-buffered LDS; every wave then runs 16 WMMAs for S
// and 16 WMMAs for P@V out of LDS. Online softmax with skip-rescale.
// ---------------------------------------------------------------------------
__global__ __launch_bounds__(128) void flash_attn(const _Float16* __restrict__ Q,
                                                  const _Float16* __restrict__ K,
                                                  const _Float16* __restrict__ VT,
                                                  float* __restrict__ out) {
  __shared__ _Float16 ksm[2][32][KROW];   // 33792 B
  __shared__ _Float16 vsm[2][256][VROW];  // 40960 B
  __shared__ _Float16 pbuf[4][16][PROW];  //  5120 B

  const int tid   = threadIdx.x;
  const int lane  = tid & 31;
  const int wave  = tid >> 5;
  const int n     = lane & 15;
  const int hs    = lane >> 4;
  const int khalf = hs * 8;
  const int row0  = (blockIdx.x * 4 + wave) * 16;

  // Staging assignment (128 threads, 16 async b128 per thread per stage):
  const int skey  = tid >> 2;
  const int spart = (tid & 3) * 128;
  const int sd    = tid * 2;

  // ---- prefetch stage for kb = 0 into buffer 0 ----
  {
    const char* gk = (const char*)(K + (size_t)skey * DD) + spart;
    const unsigned lk = lds_off(&ksm[0][skey][0]) + (unsigned)spart;
#pragma unroll
    for (int j = 0; j < 8; ++j) { async_b128(lk + j * 16, gk + j * 16); }
#pragma unroll
    for (int rr = 0; rr < 2; ++rr) {
      const char* gv = (const char*)(VT + (size_t)(sd + rr) * NN);
      const unsigned lv = lds_off(&vsm[0][sd + rr][0]);
#pragma unroll
      for (int j = 0; j < 4; ++j) { async_b128(lv + j * 16, gv + j * 16); }
    }
  }

  // ---- preload Q (16 rows x 256) as 8 A-tiles ----
  v16h qa[8];
  {
    const _Float16* qrow = Q + (size_t)(row0 + n) * DD;
#pragma unroll
    for (int s = 0; s < 8; ++s) {
      const v8h lo = *(const v8h*)(qrow + s * 32 + hs * 16);
      const v8h hi = *(const v8h*)(qrow + s * 32 + hs * 16 + 8);
#pragma unroll
      for (int j = 0; j < 8; ++j) { qa[s][j] = lo[j]; qa[s][j + 8] = hi[j]; }
    }
  }

  v8f acc[16];
#pragma unroll
  for (int t = 0; t < 16; ++t) {
#pragma unroll
    for (int r = 0; r < 8; ++r) { acc[t][r] = 0.0f; }
  }
  float mrow[8], lrow[8];
#pragma unroll
  for (int r = 0; r < 8; ++r) { mrow[r] = -__builtin_huge_valf(); lrow[r] = 0.0f; }

  int buf = 0;
  for (int kb = 0; kb < NN; kb += 32) {
    // ---- issue next stage, then wait for current stage ----
    if (kb + 32 < NN) {
      const int nb = buf ^ 1;
      const char* gk = (const char*)(K + (size_t)(kb + 32 + skey) * DD) + spart;
      const unsigned lk = lds_off(&ksm[nb][skey][0]) + (unsigned)spart;
#pragma unroll
      for (int j = 0; j < 8; ++j) { async_b128(lk + j * 16, gk + j * 16); }
#pragma unroll
      for (int rr = 0; rr < 2; ++rr) {
        const char* gv = (const char*)(VT + (size_t)(sd + rr) * NN + kb + 32);
        const unsigned lv = lds_off(&vsm[nb][sd + rr][0]);
#pragma unroll
        for (int j = 0; j < 4; ++j) { async_b128(lv + j * 16, gv + j * 16); }
      }
      asm volatile("s_wait_asynccnt 0x10" ::: "memory");
    } else {
      asm volatile("s_wait_asynccnt 0x0" ::: "memory");
    }
    __syncthreads();  // current buffer visible to all waves

    // ---- S = Q @ K^T for 32 keys (two 16x16 tiles, parallel chains) ----
    v8f s0, s1;
#pragma unroll
    for (int r = 0; r < 8; ++r) { s0[r] = 0.0f; s1[r] = 0.0f; }
    const _Float16* k0r = &ksm[buf][n][hs * 16];       // key = kb + n
    const _Float16* k1r = &ksm[buf][16 + n][hs * 16];  // key = kb + 16 + n
#pragma unroll
    for (int s = 0; s < 8; ++s) {
      v16h b0, b1;
      const v8h l0 = *(const v8h*)(k0r + s * 32);
      const v8h h0 = *(const v8h*)(k0r + s * 32 + 8);
      const v8h l1 = *(const v8h*)(k1r + s * 32);
      const v8h h1 = *(const v8h*)(k1r + s * 32 + 8);
#pragma unroll
      for (int j = 0; j < 8; ++j) {
        b0[j] = l0[j]; b0[j + 8] = h0[j];
        b1[j] = l1[j]; b1[j + 8] = h1[j];
      }
      s0 = wmma_f16(qa[s], b0, s0);
      s1 = wmma_f16(qa[s], b1, s1);
    }

    // ---- online softmax ----
    float alpha[8];
    bool upd = false;
#pragma unroll
    for (int r = 0; r < 8; ++r) {
      float mx = fmaxf(s0[r], s1[r]);
#pragma unroll
      for (int off = 1; off < 16; off <<= 1) {
        mx = fmaxf(mx, __shfl_xor(mx, off, 32));
      }
      const float mnew = fmaxf(mrow[r], mx);
      upd |= (mnew > mrow[r]);
      alpha[r] = __expf(mrow[r] - mnew);
      mrow[r] = mnew;
    }

    v8f p0, p1;
#pragma unroll
    for (int r = 0; r < 8; ++r) {
      p0[r] = __expf(s0[r] - mrow[r]);
      p1[r] = __expf(s1[r] - mrow[r]);
      float sm = p0[r] + p1[r];
#pragma unroll
      for (int off = 1; off < 16; off <<= 1) {
        sm += __shfl_xor(sm, off, 32);
      }
      lrow[r] = lrow[r] * alpha[r] + sm;
    }

    if (__ballot(upd) != 0ull) {  // rescale only when the running max moved
#pragma unroll
      for (int t = 0; t < 16; ++t) {
#pragma unroll
        for (int r = 0; r < 8; ++r) { acc[t][r] *= alpha[r]; }
      }
    }

    // ---- transpose P via per-wave LDS bounce (C layout -> A tile) ----
    _Float16* pw = &pbuf[wave][0][0];
#pragma unroll
    for (int r = 0; r < 8; ++r) {
      const int m = r + hs * 8;
      pw[m * PROW + n]      = (_Float16)p0[r];
      pw[m * PROW + 16 + n] = (_Float16)p1[r];
    }
    v16h pa;
    {
      const _Float16* pm = pw + n * PROW;
      const v8h lo = *(const v8h*)(pm + hs * 16);
      const v8h hi = *(const v8h*)(pm + hs * 16 + 8);
#pragma unroll
      for (int j = 0; j < 8; ++j) { pa[j] = lo[j]; pa[j + 8] = hi[j]; }
    }

    // ---- O += P @ V from staged V^T tile (16 independent chains) ----
#pragma unroll
    for (int t = 0; t < 16; ++t) {
      const _Float16* vr = &vsm[buf][t * 16 + n][hs * 16];
      v16h bt;
      const v8h lo = *(const v8h*)(vr);
      const v8h hi = *(const v8h*)(vr + 8);
#pragma unroll
      for (int j = 0; j < 8; ++j) { bt[j] = lo[j]; bt[j + 8] = hi[j]; }
      acc[t] = wmma_f16(pa, bt, acc[t]);
    }

    __syncthreads();  // everyone done reading this buffer before it is reused
    buf ^= 1;
  }

  // ---- epilogue: normalize and store f32 ----
#pragma unroll
  for (int r = 0; r < 8; ++r) {
    const float inv = 1.0f / lrow[r];
    const int row = row0 + r + hs * 8;
#pragma unroll
    for (int t = 0; t < 16; ++t) {
      out[(size_t)row * DD + t * 16 + n] = acc[t][r] * inv;
    }
  }
}

extern "C" void kernel_launch(void* const* d_in, const int* in_sizes, int n_in,
                              void* d_out, int out_size, void* d_ws, size_t ws_size,
                              hipStream_t stream) {
  const float* x1 = (const float*)d_in[0];
  const float* x2 = (const float*)d_in[1];
  const float* x3 = (const float*)d_in[2];
  const float* Wq = (const float*)d_in[3];
  const float* bq = (const float*)d_in[4];
  const float* Wk = (const float*)d_in[5];
  const float* bk = (const float*)d_in[6];
  const float* Wv = (const float*)d_in[7];
  const float* bv = (const float*)d_in[8];

  _Float16* q  = (_Float16*)d_ws;       // [8192][256] f16
  _Float16* k  = q + (size_t)NN * DD;   // [8192][256] f16
  _Float16* vt = k + (size_t)NN * DD;   // [256][8192] f16 (V transposed)

  dim3 qkv_grid(NN / 128, 3);  // 8 waves x 16 rows = 128 rows/block; y = q/k/v
  qkv_gemm<<<qkv_grid, 256, 0, stream>>>(x1, x2, x3, Wq, Wk, Wv, bq, bk, bv,
                                         q, k, vt);

  flash_attn<<<NN / 64, 128, 0, stream>>>(q, k, vt, (float*)d_out);
}
